// HAN_68891275428082
// MI455X (gfx1250) — matrix operand order repaired and checked by reading, not compiled, not run on hardware
//
#include <hip/hip_runtime.h>
#include <hip/hip_bf16.h>

// ---------------- problem constants (match reference) ----------------
#define NA      50000
#define NP      100000
#define INC     128
#define HIDC    128
#define OUTC    64
#define HEADS1  8
#define HEADS2  4
#define NEDGE   800000
#define SLOPE   0.2f

typedef __attribute__((ext_vector_type(16))) __bf16 v16bf;
typedef __attribute__((ext_vector_type(8)))  float  v8f;

#define CDIV(a,b) (((a)+(b)-1)/(b))

// order-preserving float <-> uint mapping so atomicMax(u32) == float max,
// and fill(0) acts as a below-(-inf) sentinel.
__device__ __forceinline__ unsigned fenc(float f) {
  unsigned u = __float_as_uint(f);
  return u ^ ((u >> 31) ? 0xFFFFFFFFu : 0x80000000u);
}
__device__ __forceinline__ float fdec(unsigned u) {
  return __uint_as_float(u ^ ((u >> 31) ? 0x80000000u : 0xFFFFFFFFu));
}

// ---------------- WMMA GEMM core ----------------
// Each wave computes a 16(M) x 32(N) output block: two 16x16 WMMA tiles that
// share one A fragment per K-step (halves A traffic + f32->bf16 cvt work).
// Fully unrolled over K (template) so 2*K/32 v_wmma ops issue back-to-back.
//
// A fragment (16x32 bf16): lane<16 -> K {kb..kb+7, kb+16..kb+23},
//                          lane>=16 -> K {kb+8..kb+15, kb+24..kb+31}.
// B fragment (32x16 bf16): lane holds W column (lane&15), K rows same as A fragment.
__device__ __forceinline__ v16bf load_b_frag(const float* __restrict__ bp, int ld) {
  v16bf B;
#pragma unroll
  for (int s = 0; s < 8; ++s) {
    B[s]     = (__bf16)bp[(size_t)s * ld];
    B[s + 8] = (__bf16)bp[(size_t)(s + 16) * ld];
  }
  return B;
}

template <int K>
__device__ __forceinline__ void wmma_tile2_f32(const float* __restrict__ X,
                                               const float* __restrict__ W,
                                               int arow, int colbase,
                                               int Fc, int lane,
                                               v8f& acc0, v8f& acc1) {
  const int half = lane >> 4;
  const int l15  = lane & 15;
  acc0 = (v8f){};
  acc1 = (v8f){};
#pragma unroll
  for (int kb = 0; kb < K; kb += 32) {
    const float* ap = X + (size_t)arow * K + (kb + half * 8);
    float4 a0 = *(const float4*)(ap);
    float4 a1 = *(const float4*)(ap + 4);
    float4 a2 = *(const float4*)(ap + 16);
    float4 a3 = *(const float4*)(ap + 20);
    v16bf A;
    A[0]=(__bf16)a0.x;  A[1]=(__bf16)a0.y;  A[2]=(__bf16)a0.z;  A[3]=(__bf16)a0.w;
    A[4]=(__bf16)a1.x;  A[5]=(__bf16)a1.y;  A[6]=(__bf16)a1.z;  A[7]=(__bf16)a1.w;
    A[8]=(__bf16)a2.x;  A[9]=(__bf16)a2.y;  A[10]=(__bf16)a2.z; A[11]=(__bf16)a2.w;
    A[12]=(__bf16)a3.x; A[13]=(__bf16)a3.y; A[14]=(__bf16)a3.z; A[15]=(__bf16)a3.w;

    const float* brow = W + (size_t)(kb + half * 8) * Fc + colbase + l15;
    v16bf B0 = load_b_frag(brow, Fc);
    v16bf B1 = load_b_frag(brow + 16, Fc);

    acc0 = __builtin_amdgcn_wmma_f32_16x16x32_bf16(false, A, false, B0,
                                                   (short)0, acc0, false, false);
    acc1 = __builtin_amdgcn_wmma_f32_16x16x32_bf16(false, A, false, B1,
                                                   (short)0, acc1, false, false);
  }
}

// C[N,Fc] = X[N,K] @ W[K,Fc] + bias.  N%16==0, Fc%32==0.
template <int K>
__global__ void HAN_gemm_bias(const float* __restrict__ X, const float* __restrict__ W,
                              const float* __restrict__ bias, float* __restrict__ C,
                              int Nrows, int Fc) {
  const int lane = threadIdx.x & 31;
  const int wid  = blockIdx.x * (blockDim.x >> 5) + (threadIdx.x >> 5);
  const int tilesN = Fc >> 5;
  if (wid >= (Nrows >> 4) * tilesN) return;
  const int rowT = wid / tilesN, colT = wid % tilesN;
  const int half = lane >> 4, l15 = lane & 15;
  const int col0 = colT * 32 + l15;

  v8f acc0, acc1;
  wmma_tile2_f32<K>(X, W, rowT * 16 + l15, colT * 32, Fc, lane, acc0, acc1);

  const float bv0 = bias[col0];
  const float bv1 = bias[col0 + 16];
  // C/D layout: lane -> column l15 (+colbase); VGPR r -> row rowT*16 + r + 8*half
  float* cp = C + (size_t)(rowT * 16 + half * 8) * Fc + col0;
#pragma unroll
  for (int r = 0; r < 8; ++r) {
    cp[(size_t)r * Fc]      = acc0[r] + bv0;
    cp[(size_t)r * Fc + 16] = acc1[r] + bv1;
  }
}

// colsum[f] += sum_n tanh( (X@W + bias)[n,f] )   (fused semantic-attention reduce)
template <int K>
__global__ void HAN_gemm_tanh_colsum(const float* __restrict__ X, const float* __restrict__ W,
                                     const float* __restrict__ bias, float* __restrict__ colsum,
                                     int Nrows, int Fc) {
  const int lane = threadIdx.x & 31;
  const int wid  = blockIdx.x * (blockDim.x >> 5) + (threadIdx.x >> 5);
  const int tilesN = Fc >> 5;
  if (wid >= (Nrows >> 4) * tilesN) return;
  const int rowT = wid / tilesN, colT = wid % tilesN;
  const int half = lane >> 4, l15 = lane & 15;
  const int col0 = colT * 32 + l15;

  v8f acc0, acc1;
  wmma_tile2_f32<K>(X, W, rowT * 16 + l15, colT * 32, Fc, lane, acc0, acc1);

  const float bv0 = bias[col0];
  const float bv1 = bias[col0 + 16];
  float s0 = 0.f, s1 = 0.f;
#pragma unroll
  for (int r = 0; r < 8; ++r) {
    s0 += tanhf(acc0[r] + bv0);
    s1 += tanhf(acc1[r] + bv1);
  }
  s0 += __shfl_down(s0, 16, 32);         // lane L + lane L+16 share a column
  s1 += __shfl_down(s1, 16, 32);
  if (half == 0) {
    atomicAdd(&colsum[col0], s0);
    atomicAdd(&colsum[col0 + 16], s1);
  }
}

// ---------------- edge attention kernels ----------------
// per-node head scores: s[n,h] = sum_d x[n, h*16+d] * a[h*16+d]   (D == 16)
__global__ void HAN_node_scores(const float* __restrict__ x, const float* __restrict__ a,
                                float* __restrict__ s, int N, int H) {
  int i = blockIdx.x * blockDim.x + threadIdx.x;
  if (i >= N * H) return;
  int n = i / H, h = i % H;
  const float* xp = x + (size_t)n * (H * 16) + h * 16;
  const float* ap = a + h * 16;
  float d = 0.f;
#pragma unroll
  for (int j = 0; j < 16; ++j) d += xp[j] * ap[j];
  s[i] = d;
}

__global__ void HAN_edge_max(const int* __restrict__ src, const int* __restrict__ dst,
                             const float* __restrict__ sS, const float* __restrict__ sD,
                             unsigned* __restrict__ m, int E, int H) {
  int e = blockIdx.x * blockDim.x + threadIdx.x;
  if (e >= E) return;
  int s0 = src[e] * H, d0 = dst[e] * H;
  for (int h = 0; h < H; ++h) {
    float v = sS[s0 + h] + sD[d0 + h];
    v = v > 0.f ? v : SLOPE * v;
    atomicMax(&m[d0 + h], fenc(v));
  }
}

__global__ void HAN_edge_expsum(const int* __restrict__ src, const int* __restrict__ dst,
                                const float* __restrict__ sS, const float* __restrict__ sD,
                                const unsigned* __restrict__ m, float* __restrict__ den,
                                int E, int H) {
  int e = blockIdx.x * blockDim.x + threadIdx.x;
  if (e >= E) return;
  int s0 = src[e] * H, d0 = dst[e] * H;
  for (int h = 0; h < H; ++h) {
    float v = sS[s0 + h] + sD[d0 + h];
    v = v > 0.f ? v : SLOPE * v;
    atomicAdd(&den[d0 + h], __expf(v - fdec(m[d0 + h])));
  }
}

// one wave per edge; lane covers Fc/32 contiguous channels (same head, D==16)
__global__ void HAN_edge_aggregate(const int* __restrict__ src, const int* __restrict__ dst,
                                   const float* __restrict__ sS, const float* __restrict__ sD,
                                   const unsigned* __restrict__ m, const float* __restrict__ den,
                                   const float* __restrict__ xsrc, float* __restrict__ out,
                                   int E, int H, int Fc) {
  int gid = blockIdx.x * blockDim.x + threadIdx.x;
  int e = gid >> 5, lane = gid & 31;
  if (e >= E) return;
  int s = src[e], d = dst[e];
  int cpl = Fc >> 5;
  int c0 = lane * cpl;
  int h = c0 >> 4;  // D == 16
  float v = sS[s * H + h] + sD[d * H + h];
  v = v > 0.f ? v : SLOPE * v;
  float p = __expf(v - fdec(m[d * H + h])) / (den[d * H + h] + 1e-16f);
  const float* xp = xsrc + (size_t)s * Fc + c0;
  float* op = out + (size_t)d * Fc + c0;
  for (int j = 0; j < cpl; ++j) atomicAdd(&op[j], p * xp[j]);
}

// ---------------- small utility kernels ----------------
__global__ void HAN_fill_u32(unsigned* __restrict__ p, unsigned v, size_t n) {
  size_t i = (size_t)blockIdx.x * blockDim.x + threadIdx.x;
  if (i < n) p[i] = v;
}
__global__ void HAN_relu(float* __restrict__ x, size_t n) {
  size_t i = (size_t)blockIdx.x * blockDim.x + threadIdx.x;
  if (i < n) x[i] = fmaxf(x[i], 0.f);
}
// semantic-attention weights over 2 metapaths
__global__ void HAN_beta2(const float* __restrict__ colsum, const float* __restrict__ q,
                          int Fc, float invN, float* __restrict__ beta) {
  float w0 = 0.f, w1 = 0.f;
  for (int f = 0; f < Fc; ++f) {
    w0 += q[f] * colsum[f] * invN;
    w1 += q[f] * colsum[Fc + f] * invN;
  }
  float mx = fmaxf(w0, w1);
  float e0 = __expf(w0 - mx), e1 = __expf(w1 - mx);
  float s = e0 + e1;
  beta[0] = e0 / s;
  beta[1] = e1 / s;
}
__global__ void HAN_combine2(const float* __restrict__ x0, const float* __restrict__ x1,
                             const float* __restrict__ beta, float* __restrict__ out, size_t n) {
  size_t i = (size_t)blockIdx.x * blockDim.x + threadIdx.x;
  if (i < n) out[i] = beta[0] * x0[i] + beta[1] * x1[i];
}

// ---------------- host orchestration ----------------
extern "C" void kernel_launch(void* const* d_in, const int* in_sizes, int n_in,
                              void* d_out, int out_size, void* d_ws, size_t ws_size,
                              hipStream_t stream) {
  (void)in_sizes; (void)n_in; (void)out_size; (void)ws_size;
  const float* x_a   = (const float*)d_in[0];
  const float* x_p   = (const float*)d_in[1];
  const float* W1_a  = (const float*)d_in[2];
  const float* b1_a  = (const float*)d_in[3];
  const float* W1_p  = (const float*)d_in[4];
  const float* b1_p  = (const float*)d_in[5];
  const float* a1s_ap = (const float*)d_in[6];
  const float* a1d_ap = (const float*)d_in[7];
  const float* a1s_pa = (const float*)d_in[8];
  const float* a1d_pa = (const float*)d_in[9];
  const float* a1s_aa = (const float*)d_in[10];
  const float* a1d_aa = (const float*)d_in[11];
  const float* q1    = (const float*)d_in[12];
  const float* Wk1   = (const float*)d_in[13];
  const float* bk1   = (const float*)d_in[14];
  const float* W2_a  = (const float*)d_in[15];
  const float* b2_a  = (const float*)d_in[16];
  const float* W2_p  = (const float*)d_in[17];
  const float* b2_p  = (const float*)d_in[18];
  const float* a2s_pa = (const float*)d_in[21];
  const float* a2d_pa = (const float*)d_in[22];
  const float* a2s_aa = (const float*)d_in[23];
  const float* a2d_aa = (const float*)d_in[24];
  const float* q2    = (const float*)d_in[25];
  const float* Wk2   = (const float*)d_in[26];
  const float* bk2   = (const float*)d_in[27];
  const int* ei_ap_s = (const int*)d_in[28];
  const int* ei_ap_d = (const int*)d_in[29];
  const int* ei_pa_s = (const int*)d_in[30];
  const int* ei_pa_d = (const int*)d_in[31];
  const int* ei_aa_s = (const int*)d_in[32];
  const int* ei_aa_d = (const int*)d_in[33];
  float* outp = (float*)d_out;

  // workspace layout (floats)
  float* ws = (float*)d_ws;
  float* ha1   = ws;                         // NA*128
  float* hp1   = ha1  + (size_t)NA * HIDC;   // NP*128
  float* o_ap  = hp1  + (size_t)NP * HIDC;   // NP*128 (becomes p1)
  float* o_pa  = o_ap + (size_t)NP * HIDC;   // NA*128
  float* o_aa  = o_pa + (size_t)NA * HIDC;   // NA*128
  float* a1v   = o_aa + (size_t)NA * HIDC;   // NA*128
  float* ha2   = a1v  + (size_t)NA * HIDC;   // NA*64
  float* hp2   = ha2  + (size_t)NA * OUTC;   // NP*64
  float* o_pa2 = hp2  + (size_t)NP * OUTC;   // NA*64
  float* o_aa2 = o_pa2 + (size_t)NA * OUTC;  // NA*64
  float* sS    = o_aa2 + (size_t)NA * OUTC;  // <= NP*8
  float* sD    = sS  + (size_t)NP * HEADS1;  // <= NP*8
  float* den   = sD  + (size_t)NP * HEADS1;  // <= NP*8
  unsigned* menc = (unsigned*)(den + (size_t)NP * HEADS1); // <= NP*8
  float* colsum = (float*)(menc + (size_t)NP * HEADS1);    // 2*128
  float* beta   = colsum + 2 * HIDC;                       // 2

  const int TB = 256;                 // threads per block
  const int WPB = TB / 32;            // waves per block

  auto gemmBlocks = [&](int n, int fc) { return CDIV((n / 16) * (fc / 32), WPB); };

  // ---- layer 1 projections (WMMA, K=128) ----
  HAN_gemm_bias<INC><<<gemmBlocks(NA, HIDC), TB, 0, stream>>>(x_a, W1_a, b1_a, ha1, NA, HIDC);
  HAN_gemm_bias<INC><<<gemmBlocks(NP, HIDC), TB, 0, stream>>>(x_p, W1_p, b1_p, hp1, NP, HIDC);

  // ---- edge type pipeline ----
  auto edge_type = [&](const float* xs, const float* xd, const float* as, const float* ad,
                       const int* es, const int* ed, int Nsrc, int Ndst,
                       float* out, int H, int Fc) {
    HAN_node_scores<<<CDIV(Nsrc * H, TB), TB, 0, stream>>>(xs, as, sS, Nsrc, H);
    HAN_node_scores<<<CDIV(Ndst * H, TB), TB, 0, stream>>>(xd, ad, sD, Ndst, H);
    size_t nh = (size_t)Ndst * H, nf = (size_t)Ndst * Fc;
    HAN_fill_u32<<<CDIV(nh, (size_t)TB), TB, 0, stream>>>(menc, 0u, nh);            // -inf sentinel
    HAN_fill_u32<<<CDIV(nh, (size_t)TB), TB, 0, stream>>>((unsigned*)den, 0u, nh);  // 0.0f
    HAN_fill_u32<<<CDIV(nf, (size_t)TB), TB, 0, stream>>>((unsigned*)out, 0u, nf);  // 0.0f
    HAN_edge_max<<<CDIV(NEDGE, TB), TB, 0, stream>>>(es, ed, sS, sD, menc, NEDGE, H);
    HAN_edge_expsum<<<CDIV(NEDGE, TB), TB, 0, stream>>>(es, ed, sS, sD, menc, den, NEDGE, H);
    HAN_edge_aggregate<<<CDIV((size_t)NEDGE * 32, (size_t)TB), TB, 0, stream>>>(
        es, ed, sS, sD, menc, den, xs, out, NEDGE, H, Fc);
    HAN_relu<<<CDIV(nf, (size_t)TB), TB, 0, stream>>>(out, nf);
  };

  // layer 1: author->paper, paper->author, author->author
  edge_type(ha1, hp1, a1s_ap, a1d_ap, ei_ap_s, ei_ap_d, NA, NP, o_ap, HEADS1, HIDC);
  edge_type(hp1, ha1, a1s_pa, a1d_pa, ei_pa_s, ei_pa_d, NP, NA, o_pa, HEADS1, HIDC);
  edge_type(ha1, ha1, a1s_aa, a1d_aa, ei_aa_s, ei_aa_d, NA, NA, o_aa, HEADS1, HIDC);

  // ---- layer 1 semantic attention for authors (papers: single metapath -> p1 == o_ap) ----
  HAN_fill_u32<<<1, 256, 0, stream>>>((unsigned*)colsum, 0u, 2 * HIDC);
  HAN_gemm_tanh_colsum<HIDC><<<gemmBlocks(NA, HIDC), TB, 0, stream>>>(o_pa, Wk1, bk1, colsum, NA, HIDC);
  HAN_gemm_tanh_colsum<HIDC><<<gemmBlocks(NA, HIDC), TB, 0, stream>>>(o_aa, Wk1, bk1, colsum + HIDC, NA, HIDC);
  HAN_beta2<<<1, 1, 0, stream>>>(colsum, q1, HIDC, 1.f / NA, beta);
  HAN_combine2<<<CDIV((size_t)NA * HIDC, (size_t)TB), TB, 0, stream>>>(o_pa, o_aa, beta, a1v,
                                                                       (size_t)NA * HIDC);

  // ---- layer 2 projections (WMMA, K=128); ap edge type feeds only the discarded paper output ----
  HAN_gemm_bias<HIDC><<<gemmBlocks(NA, OUTC), TB, 0, stream>>>(a1v, W2_a, b2_a, ha2, NA, OUTC);
  HAN_gemm_bias<HIDC><<<gemmBlocks(NP, OUTC), TB, 0, stream>>>(o_ap, W2_p, b2_p, hp2, NP, OUTC);

  edge_type(hp2, ha2, a2s_pa, a2d_pa, ei_pa_s, ei_pa_d, NP, NA, o_pa2, HEADS2, OUTC);
  edge_type(ha2, ha2, a2s_aa, a2d_aa, ei_aa_s, ei_aa_d, NA, NA, o_aa2, HEADS2, OUTC);

  // ---- layer 2 semantic attention (K=64) -> final author features ----
  HAN_fill_u32<<<1, 256, 0, stream>>>((unsigned*)colsum, 0u, 2 * OUTC);
  HAN_gemm_tanh_colsum<OUTC><<<gemmBlocks(NA, OUTC), TB, 0, stream>>>(o_pa2, Wk2, bk2, colsum, NA, OUTC);
  HAN_gemm_tanh_colsum<OUTC><<<gemmBlocks(NA, OUTC), TB, 0, stream>>>(o_aa2, Wk2, bk2, colsum + OUTC, NA, OUTC);
  HAN_beta2<<<1, 1, 0, stream>>>(colsum, q2, OUTC, 1.f / NA, beta);
  HAN_combine2<<<CDIV((size_t)NA * OUTC, (size_t)TB), TB, 0, stream>>>(o_pa2, o_aa2, beta, outp,
                                                                       (size_t)NA * OUTC);
}
